// QuerySAT_34849364640213
// MI455X (gfx1250) — matrix operand order repaired and checked by reading, not compiled, not run on hardware
//
#include <hip/hip_runtime.h>

// ---------------------------------------------------------------------------
// Problem constants (from the reference)
// ---------------------------------------------------------------------------
#define GNUM      32
#define NVARS     16384
#define NCLAUSES  32768
#define NLITS     98304
#define FMDIM     128
#define QMDIM     32
#define NROUNDS   4

typedef __attribute__((ext_vector_type(16))) __bf16        v16bf;
typedef __attribute__((ext_vector_type(8)))  float         v8f;
typedef __attribute__((ext_vector_type(4)))  unsigned int  uint32x4;
typedef __attribute__((ext_vector_type(8)))  int           int32x8;
typedef __attribute__((ext_vector_type(4)))  int           int32x4;

// Tensor Data Mover availability (device pass only; host pass parses fallback)
#if defined(__HIP_DEVICE_COMPILE__)
#  if defined(__has_builtin)
#    if __has_builtin(__builtin_amdgcn_tensor_load_to_lds) && \
        __has_builtin(__builtin_amdgcn_s_wait_tensorcnt)
#      define QS_TDM 1
#    endif
#  endif
#endif

static __device__ __forceinline__ float softplus_f(float x) {
    if (x > 20.f)  return x;
    if (x < -20.f) return expf(x);
    return log1pf(expf(x));
}

// ---------------------------------------------------------------------------
// Tiled WMMA GEMM:  Y[N,F] = act(X[N,K] @ W[K,F] + bias)
// X bf16 row-major (K mult of 32), W bf16 row-major, bias f32.
// Block = 128 threads (4 waves); block tile M=128, N=32.
// Each wave computes a 32x32 C tile: 2 A frags x 2 B frags -> 4 WMMA / K-step.
// X tile staged global->LDS by the TDM (async tensor DMA) when available,
// using TDM LDS padding (16 dwords + 4 pad dwords) to realize row stride 40.
// ---------------------------------------------------------------------------
template <bool ACT>
__global__ __launch_bounds__(128) void k_gemm(const __bf16* __restrict__ X,
                                              const __bf16* __restrict__ W,
                                              const float* __restrict__ bias,
                                              void* __restrict__ Yv,
                                              int K, int F) {
    __shared__ __align__(16) __bf16 Xs[128][40];  // A tile, row-major, stride 40
    __shared__ __align__(16) __bf16 Wts[32][40];  // B tile, transposed [n][k]

    const int t      = threadIdx.x;
    const int lane   = t & 31;
    const int wave   = t >> 5;          // 0..3
    const int mBase  = wave * 32;
    const int rowBlk = blockIdx.x * 128;
    const int colBlk = blockIdx.y * 32;
    const int half   = lane >> 4, l15 = lane & 15;

    v8f acc00 = {0.f,0.f,0.f,0.f,0.f,0.f,0.f,0.f};
    v8f acc01 = acc00, acc10 = acc00, acc11 = acc00;

    const int wr = t >> 2, wc = (t & 3) * 8;    // W loader: 32 rows x 32 cols
    const int xc = (t & 3) * 8;                  // X fallback loader col

    for (int k0 = 0; k0 < K; k0 += 32) {
        __syncthreads();

        // ---- stage W tile transposed: rows k0..k0+31, cols colBlk..+31 ----
        {
            union { uint4 u; __bf16 h[8]; } wv;
            wv.u = *reinterpret_cast<const uint4*>(W + (size_t)(k0 + wr) * F + colBlk + wc);
#pragma unroll
            for (int i = 0; i < 8; ++i) Wts[wc + i][wr] = wv.h[i];
        }

        // ---- stage X tile: 128 rows x 32 cols ----
#if defined(QS_TDM)
        if (wave == 0) {
            unsigned long long ga = (unsigned long long)(size_t)X +
                                    ((unsigned long long)(size_t)rowBlk * (unsigned)K +
                                     (unsigned)k0) * 2ull;
            unsigned ldsA = (unsigned)(size_t)(&Xs[0][0]);
            uint32x4 g0 = {0u, 0u, 0u, 0u};
            g0[0] = 1u;                                        // count = 1 valid D#
            g0[1] = ldsA;                                      // lds_addr (bytes)
            g0[2] = (unsigned)(ga & 0xFFFFFFFFu);              // global_addr[31:0]
            g0[3] = (unsigned)((ga >> 32) & 0x01FFFFFFu)       // global_addr[56:32]
                    | (2u << 30);                              // type = 2 (image)
            int32x8 g1 = {0, 0, 0, 0, 0, 0, 0, 0};
            g1[0] = (int)((1u << 16)      // data_size = 1 -> 2 bytes
                  |       (1u << 20)      // pad_enable
                  |       (3u << 22)      // pad_interval: 16 dwords
                  |       (3u << 25));    // pad_amount:   4 dwords (stride 40 bf16)
            g1[1] = (int)(((unsigned)K & 0xFFFFu) << 16);      // tensor_dim0 = K
            g1[2] = (int)(0x8000u << 16);                      // tensor_dim1 = 32768
            g1[3] = (int)(32u << 16);                          // tile_dim0 = 32
            g1[4] = 128;                                       // tile_dim1 = 128
            g1[5] = K;                                         // tensor_dim0_stride
            int32x4 gz = {0, 0, 0, 0};
#if __clang_major__ >= 23
            int32x8 gz8 = {0, 0, 0, 0, 0, 0, 0, 0};
            __builtin_amdgcn_tensor_load_to_lds(g0, g1, gz, gz, gz8, 0);
#else
            __builtin_amdgcn_tensor_load_to_lds(g0, g1, gz, gz, 0);
#endif
            __builtin_amdgcn_s_wait_tensorcnt(0);
        }
#else
#pragma unroll
        for (int p = 0; p < 4; ++p) {
            int r = p * 32 + (t >> 2);
            *reinterpret_cast<uint4*>(&Xs[r][xc]) =
                *reinterpret_cast<const uint4*>(X + (size_t)(rowBlk + r) * K + k0 + xc);
        }
#endif
        __syncthreads();

        // ---- fragments ----
        // A: lane(0..15)=row M, elems 0..7 -> K=half*8+i, 8..15 -> K=16+half*8+i
        v16bf a0, a1, b0, b1;
        reinterpret_cast<uint4*>(&a0)[0] =
            *reinterpret_cast<const uint4*>(&Xs[mBase + l15][half * 8]);
        reinterpret_cast<uint4*>(&a0)[1] =
            *reinterpret_cast<const uint4*>(&Xs[mBase + l15][16 + half * 8]);
        reinterpret_cast<uint4*>(&a1)[0] =
            *reinterpret_cast<const uint4*>(&Xs[mBase + 16 + l15][half * 8]);
        reinterpret_cast<uint4*>(&a1)[1] =
            *reinterpret_cast<const uint4*>(&Xs[mBase + 16 + l15][16 + half * 8]);
        // B: lane(0..15)=col N, elems 0..7 -> K=half*8+i, 8..15 -> K=16+half*8+i
        reinterpret_cast<uint4*>(&b0)[0] =
            *reinterpret_cast<const uint4*>(&Wts[l15][half * 8]);
        reinterpret_cast<uint4*>(&b0)[1] =
            *reinterpret_cast<const uint4*>(&Wts[l15][16 + half * 8]);
        reinterpret_cast<uint4*>(&b1)[0] =
            *reinterpret_cast<const uint4*>(&Wts[16 + l15][half * 8]);
        reinterpret_cast<uint4*>(&b1)[1] =
            *reinterpret_cast<const uint4*>(&Wts[16 + l15][16 + half * 8]);

        acc00 = __builtin_amdgcn_wmma_f32_16x16x32_bf16(false, a0, false, b0, (short)0, acc00, false, false);
        acc01 = __builtin_amdgcn_wmma_f32_16x16x32_bf16(false, a0, false, b1, (short)0, acc01, false, false);
        acc10 = __builtin_amdgcn_wmma_f32_16x16x32_bf16(false, a1, false, b0, (short)0, acc10, false, false);
        acc11 = __builtin_amdgcn_wmma_f32_16x16x32_bf16(false, a1, false, b1, (short)0, acc11, false, false);
    }

    // C/D layout: lanes 0-15 VGPR r -> M=r; lanes 16-31 -> M=8+r; N = lane&15
    const int col0 = colBlk + l15;
    const int col1 = colBlk + 16 + l15;
    const float bv0 = bias[col0], bv1 = bias[col1];
    const int rA = rowBlk + mBase + half * 8;
    const int rB = rA + 16;
#pragma unroll
    for (int r = 0; r < 8; ++r) {
        float y00 = acc00[r] + bv0, y01 = acc01[r] + bv1;
        float y10 = acc10[r] + bv0, y11 = acc11[r] + bv1;
        if (ACT) {
            y00 = y00 > 0.f ? y00 : 0.2f * y00;
            y01 = y01 > 0.f ? y01 : 0.2f * y01;
            y10 = y10 > 0.f ? y10 : 0.2f * y10;
            y11 = y11 > 0.f ? y11 : 0.2f * y11;
            __bf16* Y = reinterpret_cast<__bf16*>(Yv);
            Y[(size_t)(rA + r) * F + col0] = (__bf16)y00;
            Y[(size_t)(rA + r) * F + col1] = (__bf16)y01;
            Y[(size_t)(rB + r) * F + col0] = (__bf16)y10;
            Y[(size_t)(rB + r) * F + col1] = (__bf16)y11;
        } else {
            float* Y = reinterpret_cast<float*>(Yv);
            Y[(size_t)(rA + r) * F + col0] = y00;
            Y[(size_t)(rA + r) * F + col1] = y01;
            Y[(size_t)(rB + r) * F + col0] = y10;
            Y[(size_t)(rB + r) * F + col1] = y11;
        }
    }
}

// ---------------------------------------------------------------------------
// Utility / elementwise kernels
// ---------------------------------------------------------------------------
__global__ void k_zero(float* p, int n) {
    int t = blockIdx.x * blockDim.x + threadIdx.x;
    if (t < n) p[t] = 0.f;
}

// _zero_state: col0 = (1-1/128)*sqrt(128)*0.25 ; else -1/128*sqrt(128)*0.25
__global__ void k_init_state(float* p, int n) {
    int t = blockIdx.x * blockDim.x + threadIdx.x;
    if (t >= n) return;
    const float A = 2.8284271247f * (127.f / 128.f);
    const float B = -2.8284271247f / 128.f;
    p[t] = ((t & 127) == 0) ? A : B;
}

// fp32 weight (fin x fout) -> bf16 padded (kpad x fout), zero pad rows
__global__ void k_w2bf(const float* __restrict__ w, __bf16* __restrict__ dst,
                       int fin, int fout, int kpad) {
    int t = blockIdx.x * blockDim.x + threadIdx.x;
    if (t >= kpad * fout) return;
    int k = t / fout;
    dst[t] = (k < fin) ? (__bf16)w[t] : (__bf16)0.f;
}

// v1 = [variables(128) | noise_t(4) | zeros(28)] in bf16, stride 160
__global__ void k_build_v1(const float* __restrict__ vars,
                           const float* __restrict__ noise_t,
                           __bf16* __restrict__ v1) {
    int row = blockIdx.x, j = threadIdx.x;  // blockDim = 160
    float v;
    if (j < 128)      v = vars[(size_t)row * 128 + j];
    else if (j < 132) v = noise_t[(size_t)row * 4 + (j - 128)];
    else              v = 0.f;
    v1[(size_t)row * 160 + j] = (__bf16)v;
}

// cm_in cols 0..127 = clause_state (bf16); cols 128..159 filled by k_clause_grad
__global__ void k_build_cm_in(const float* __restrict__ cs, __bf16* __restrict__ cm_in) {
    int row = blockIdx.x, j = threadIdx.x;  // blockDim = 128
    cm_in[(size_t)row * 160 + j] = (__bf16)cs[(size_t)row * 128 + j];
}

// Per (clause, feature): cl = exp(-sum_j softplus(q[v_j]*s_j)); write to cm_in;
// grad wrt q: -cl * sigmoid(q*s)*s scattered into vgrad (f32 atomics).
__global__ void k_clause_grad(const float* __restrict__ query,
                              const int* __restrict__ lit_var,
                              const int* __restrict__ lit_sign,
                              __bf16* __restrict__ cm_in,
                              float* __restrict__ vgrad) {
    int t = blockIdx.x * blockDim.x + threadIdx.x;  // NCLAUSES*QM threads
    int f = t & 31, c = t >> 5;
    float s = 0.f, sig[3], sg[3];
    int vs[3];
#pragma unroll
    for (int j = 0; j < 3; ++j) {
        int l = 3 * c + j;
        int v = lit_var[l];
        float g = (float)lit_sign[l];
        float x = query[(size_t)v * QMDIM + f] * g;
        s += softplus_f(x);
        sig[j] = 1.f / (1.f + expf(-x));
        vs[j] = v; sg[j] = g;
    }
    float cl = expf(-s);
    cm_in[(size_t)c * 160 + 128 + f] = (__bf16)cl;
#pragma unroll
    for (int j = 0; j < 3; ++j)
        atomicAdd(&vgrad[(size_t)vs[j] * QMDIM + f], -cl * sig[j] * sg[j]);
}

// Per (literal, feature): contrib = vloss[clause][f] -> loss_pos/neg[var][f]
__global__ void k_scatter_loss(const float* __restrict__ clause_data,
                               const int* __restrict__ lit_clause,
                               const int* __restrict__ lit_var,
                               const int* __restrict__ lit_sign,
                               float* __restrict__ loss_pos,
                               float* __restrict__ loss_neg) {
    int t = blockIdx.x * blockDim.x + threadIdx.x;  // NLITS*QM threads
    int f = t & 31, l = t >> 5;
    int c = lit_clause[l];
    int v = lit_var[l];
    float contrib = clause_data[(size_t)c * 160 + f];  // vloss = first 32 cols
    float* dst = (lit_sign[l] > 0) ? loss_pos : loss_neg;
    atomicAdd(&dst[(size_t)v * QMDIM + f], contrib);
}

// unit = [variables(128) | vgrad(32) | loss_pos(32) | loss_neg(32)] bf16, stride 224
__global__ void k_build_ug_in(const float* __restrict__ vars,
                              const float* __restrict__ vgrad,
                              const float* __restrict__ lp,
                              const float* __restrict__ ln,
                              __bf16* __restrict__ ug_in) {
    int row = blockIdx.x, j = threadIdx.x;  // blockDim = 224
    float v;
    if (j < 128)      v = vars[(size_t)row * 128 + j];
    else if (j < 160) v = vgrad[(size_t)row * 32 + (j - 128)];
    else if (j < 192) v = lp[(size_t)row * 32 + (j - 160)];
    else              v = ln[(size_t)row * 32 + (j - 192)];
    ug_in[(size_t)row * 224 + j] = (__bf16)v;
}

// Segment sums for pair_norm mean (128 features). Block = 128 thr, 256 rows.
__global__ void k_segsum(const float* __restrict__ x, float* __restrict__ gsum,
                         int stride, int off, int rowsPerGraph) {
    int f = threadIdx.x;
    int r0 = blockIdx.x * 256;
    int g = r0 / rowsPerGraph;
    float s = 0.f;
    for (int r = 0; r < 256; ++r) s += x[(size_t)(r0 + r) * stride + off + f];
    atomicAdd(&gsum[g * 128 + f], s);
}

// pair_norm apply + residual: state = 0.25*norm(x - mean) + 0.1*state
// Optionally emits bf16 copy of the new state (input to next GEMM).
// Block = 256 thr = 8 waves, one row per wave, 4 features per lane.
__global__ void k_pairnorm(const float* __restrict__ x, const float* __restrict__ gsum,
                           float* __restrict__ state, __bf16* __restrict__ bfout,
                           int stride, int off, int rowsPerGraph, float invCnt) {
    int row = blockIdx.x * 8 + (threadIdx.x >> 5);
    int lane = threadIdx.x & 31;
    int g = row / rowsPerGraph;
    float xv[4], ss = 0.f;
#pragma unroll
    for (int i = 0; i < 4; ++i) {
        int f = lane + 32 * i;
        float v = x[(size_t)row * stride + off + f] - gsum[g * 128 + f] * invCnt;
        xv[i] = v; ss += v * v;
    }
    for (int m = 16; m; m >>= 1) ss += __shfl_xor(ss, m, 32);
    float rms = rsqrtf(ss * (1.f / 128.f) + 1e-6f);
#pragma unroll
    for (int i = 0; i < 4; ++i) {
        int f = lane + 32 * i;
        size_t idx = (size_t)row * 128 + f;
        float nv = 0.25f * xv[i] * rms + 0.1f * state[idx];
        state[idx] = nv;
        if (bfout) bfout[idx] = (__bf16)nv;
    }
}

// Final vo layer: logits = h2(128) . w2 + b2. One wave per row.
__global__ void k_vo_final(const __bf16* __restrict__ h, const float* __restrict__ w,
                           const float* __restrict__ b, float* __restrict__ out) {
    int row = blockIdx.x * 8 + (threadIdx.x >> 5);
    int lane = threadIdx.x & 31;
    float s = 0.f;
    for (int i = lane; i < 128; i += 32) s += (float)h[(size_t)row * 128 + i] * w[i];
    for (int m = 16; m; m >>= 1) s += __shfl_xor(s, m, 32);
    if (lane == 0) out[row] = s + b[0];
}

// Per clause: v = exp(-sum softplus(logit*s)); pc = v * -log(1-v+eps); seg-sum per graph
__global__ void k_clause_loss(const float* __restrict__ logits,
                              const int* __restrict__ lit_var,
                              const int* __restrict__ lit_sign,
                              float* __restrict__ per_graph) {
    int c = blockIdx.x * blockDim.x + threadIdx.x;
    if (c >= NCLAUSES) return;
    float s = 0.f;
#pragma unroll
    for (int j = 0; j < 3; ++j) {
        int l = 3 * c + j;
        float x = logits[lit_var[l]] * (float)lit_sign[l];
        s += softplus_f(x);
    }
    float v = expf(-s);
    float pc = v * -logf(1.f - v + 1e-8f);
    atomicAdd(&per_graph[c >> 10], pc);
}

__global__ void k_round_loss(const float* __restrict__ per_graph, float* __restrict__ accum) {
    float x = sqrtf(per_graph[threadIdx.x] + 1e-6f);  // 32 threads, one per graph
    for (int m = 16; m; m >>= 1) x += __shfl_xor(x, m, 32);
    if (threadIdx.x == 0) atomicAdd(accum, x);
}

__global__ void k_write_loss(const float* __restrict__ accum, float* __restrict__ out) {
    out[NVARS] = accum[0] * (1.f / (float)NROUNDS);
}

// ---------------------------------------------------------------------------
// Host driver
// ---------------------------------------------------------------------------
extern "C" void kernel_launch(void* const* d_in, const int* in_sizes, int n_in,
                              void* d_out, int out_size, void* d_ws, size_t ws_size,
                              hipStream_t stream) {
    (void)in_sizes; (void)n_in; (void)out_size; (void)ws_size;

    const int*   lit_var    = (const int*)d_in[0];
    const int*   lit_sign   = (const int*)d_in[1];
    const int*   lit_clause = (const int*)d_in[2];
    const float* noise      = (const float*)d_in[5];
    float*       out        = (float*)d_out;

    // ---- workspace allocator (256B aligned slices of d_ws) ----
    size_t off = 0;
    auto alloc = [&](size_t bytes) -> void* {
        void* p = (char*)d_ws + off;
        off += (bytes + 255) & ~(size_t)255;
        return p;
    };
    auto cdiv = [](int a, int b) { return (a + b - 1) / b; };

    // ---- bf16 weights (converted every call; deterministic) ----
    const int wfin [11] = {132, 64, 64, 160, 384, 384, 224, 256, 256, 128, 128};
    const int wkpad[11] = {160, 64, 64, 160, 384, 384, 224, 256, 256, 128, 128};
    const int wfout[11] = { 64, 64, 32, 384, 384, 160, 256, 256, 128, 128, 128};
    const int widx [11] = {  6,  8, 10,  12,  14,  16,  18,  20,  22,  24,  26};
    __bf16* wb[11];
    for (int i = 0; i < 11; ++i) {
        int n = wkpad[i] * wfout[i];
        wb[i] = (__bf16*)alloc((size_t)n * 2);
        k_w2bf<<<cdiv(n, 256), 256, 0, stream>>>((const float*)d_in[widx[i]], wb[i],
                                                 wfin[i], wfout[i], wkpad[i]);
    }
    auto bias = [&](int i) { return (const float*)d_in[i]; };

    // ---- state + scratch buffers ----
    float*  variables    = (float*)alloc((size_t)NVARS * 128 * 4);
    float*  clause_state = (float*)alloc((size_t)NCLAUSES * 128 * 4);

    const int ZTOT = 524288 * 3 + 4096 * 2 + 32;          // zeroed each round
    float* zbase     = (float*)alloc((size_t)ZTOT * 4);
    float* vgrad     = zbase;
    float* loss_pos  = vgrad + 524288;
    float* loss_neg  = loss_pos + 524288;
    float* gsum_c    = loss_neg + 524288;
    float* gsum_v    = gsum_c + 4096;
    float* per_graph = gsum_v + 4096;
    float* loss_acc  = (float*)alloc(256);

    __bf16* v1_bf   = (__bf16*)alloc((size_t)NVARS * 160 * 2);
    __bf16* vq_h1   = (__bf16*)alloc((size_t)NVARS * 64 * 2);
    __bf16* vq_h2   = (__bf16*)alloc((size_t)NVARS * 64 * 2);
    float*  query   = (float*)alloc((size_t)NVARS * 32 * 4);
    __bf16* cm_in   = (__bf16*)alloc((size_t)NCLAUSES * 160 * 2);
    __bf16* cm_h1   = (__bf16*)alloc((size_t)NCLAUSES * 384 * 2);
    __bf16* cm_h2   = (__bf16*)alloc((size_t)NCLAUSES * 384 * 2);
    float*  cdata   = (float*)alloc((size_t)NCLAUSES * 160 * 4);
    __bf16* ug_in   = (__bf16*)alloc((size_t)NVARS * 224 * 2);
    __bf16* ug_h1   = (__bf16*)alloc((size_t)NVARS * 256 * 2);
    __bf16* ug_h2   = (__bf16*)alloc((size_t)NVARS * 256 * 2);
    float*  ug_out  = (float*)alloc((size_t)NVARS * 128 * 4);
    __bf16* var_bf  = (__bf16*)alloc((size_t)NVARS * 128 * 2);
    __bf16* vo_h1   = (__bf16*)alloc((size_t)NVARS * 128 * 2);
    __bf16* vo_h2   = (__bf16*)alloc((size_t)NVARS * 128 * 2);

    auto gemm = [&](const void* X, const __bf16* W, const float* b, void* Y,
                    int N, int K, int F, bool act) {
        dim3 grid(N / 128, F / 32);
        if (act)
            k_gemm<true><<<grid, 128, 0, stream>>>((const __bf16*)X, W, b, Y, K, F);
        else
            k_gemm<false><<<grid, 128, 0, stream>>>((const __bf16*)X, W, b, Y, K, F);
    };

    // ---- init ----
    k_init_state<<<cdiv(NVARS * 128, 256), 256, 0, stream>>>(variables, NVARS * 128);
    k_init_state<<<cdiv(NCLAUSES * 128, 256), 256, 0, stream>>>(clause_state, NCLAUSES * 128);
    k_zero<<<1, 32, 0, stream>>>(loss_acc, 1);

    // ---- rounds ----
    for (int t = 0; t < NROUNDS; ++t) {
        const float* noise_t = noise + (size_t)t * NVARS * 4;
        k_zero<<<cdiv(ZTOT, 256), 256, 0, stream>>>(zbase, ZTOT);

        // query = MLP_vq([variables, noise])
        k_build_v1<<<NVARS, 160, 0, stream>>>(variables, noise_t, v1_bf);
        gemm(v1_bf, wb[0], bias(7),  vq_h1, NVARS, 160, 64, true);
        gemm(vq_h1, wb[1], bias(9),  vq_h2, NVARS, 64,  64, true);
        gemm(vq_h2, wb[2], bias(11), query, NVARS, 64,  32, false);

        // clause values + variables_grad; build cm input
        k_build_cm_in<<<NCLAUSES, 128, 0, stream>>>(clause_state, cm_in);
        k_clause_grad<<<cdiv(NCLAUSES * 32, 256), 256, 0, stream>>>(
            query, lit_var, lit_sign, cm_in, vgrad);

        // clause_data = MLP_cm([clause_state, clauses_loss])
        gemm(cm_in, wb[3], bias(13), cm_h1, NCLAUSES, 160, 384, true);
        gemm(cm_h1, wb[4], bias(15), cm_h2, NCLAUSES, 384, 384, true);
        gemm(cm_h2, wb[5], bias(17), cdata, NCLAUSES, 384, 160, false);

        // clause_state = 0.25*pair_norm(clause_data[:,32:]) + 0.1*clause_state
        k_segsum<<<NCLAUSES / 256, 128, 0, stream>>>(cdata, gsum_c, 160, 32, 1024);
        k_pairnorm<<<NCLAUSES / 8, 256, 0, stream>>>(cdata, gsum_c, clause_state,
                                                     (__bf16*)nullptr,
                                                     160, 32, 1024, 1.f / 1024.f);

        // loss_pos / loss_neg scatter from vloss
        k_scatter_loss<<<cdiv(NLITS * 32, 256), 256, 0, stream>>>(
            cdata, lit_clause, lit_var, lit_sign, loss_pos, loss_neg);

        // variables = 0.25*pair_norm(MLP_ug(unit)) + 0.1*variables
        k_build_ug_in<<<NVARS, 224, 0, stream>>>(variables, vgrad, loss_pos, loss_neg, ug_in);
        gemm(ug_in, wb[6], bias(19), ug_h1, NVARS, 224, 256, true);
        gemm(ug_h1, wb[7], bias(21), ug_h2, NVARS, 256, 256, true);
        gemm(ug_h2, wb[8], bias(23), ug_out, NVARS, 256, 128, false);
        k_segsum<<<NVARS / 256, 128, 0, stream>>>(ug_out, gsum_v, 128, 0, 512);
        k_pairnorm<<<NVARS / 8, 256, 0, stream>>>(ug_out, gsum_v, variables, var_bf,
                                                  128, 0, 512, 1.f / 512.f);

        // logits = MLP_vo(variables)
        gemm(var_bf, wb[9],  bias(25), vo_h1, NVARS, 128, 128, true);
        gemm(vo_h1,  wb[10], bias(27), vo_h2, NVARS, 128, 128, true);
        k_vo_final<<<NVARS / 8, 256, 0, stream>>>(vo_h2, (const float*)d_in[28],
                                                  (const float*)d_in[29], out);

        // logit_loss accumulation
        k_clause_loss<<<NCLAUSES / 256, 256, 0, stream>>>(out, lit_var, lit_sign, per_graph);
        k_round_loss<<<1, 32, 0, stream>>>(per_graph, loss_acc);
    }

    k_write_loss<<<1, 32, 0, stream>>>(loss_acc, out);
}
